// HPNLearner_9242769621491
// MI455X (gfx1250) — matrix operand order repaired and checked by reading, not compiled, not run on hardware
//
#include <hip/hip_runtime.h>
#include <hip/hip_bf16.h>

typedef __attribute__((ext_vector_type(16))) __bf16 v16bf_t;
typedef __attribute__((ext_vector_type(8)))  float  v8f_t;

// ---------------------------------------------------------------------------
// Generic strided implicit-GEMM conv on the CDNA5 matrix cores.
// Two variants, both computing 16(cout)x16(pixel) tiles per wave with
// v_wmma_f32_16x16x32_bf16:
//  - conv_wmma_direct: 1 wave/block, B fragment built in registers (small Cout)
//  - conv_wmma_shared: 4 waves/block sharing one pixel tile; the im2col B
//    fragment (64x16 bf16 per step) is built cooperatively into double-buffered
//    LDS and read back as ds_load_b128; two WMMAs per barrier.
// Templated on conv kernel size KK so K->(ci,kh,kw) uses constant divisors.
// ---------------------------------------------------------------------------
struct ConvArgs {
  const float* x;
  const __bf16* w;     // (Cout, Kpad) row-major, K zero-padded to mult of 32
  const float* bias;
  float* y;
  int  B1, B2;
  long inB0, inB1, inB2;
  long outB0, outB1, outB2;
  int  Hin, Win;  long inH, inW;
  int  Cin;       long inC;
  int  Hout, Wout; long outH, outW;
  int  Cout;      long outC;
  int  Kk, stride, pad;
  int  Ktot, Kpad;
  int  accum, relu;
};

__device__ __forceinline__ void conv_store(const ConvArgs& a, const v8f_t& acc,
                                           long outBase, int coTile, int half,
                                           int ho, int wo) {
  if (ho >= 0) {
#pragma unroll
    for (int rr = 0; rr < 8; ++rr) {
      int co = coTile + rr + half * 8;
      if (co < a.Cout) {
        float v = acc[rr] + a.bias[co];
        long addr = outBase + (long)co * a.outC + (long)ho * a.outH + (long)wo * a.outW;
        if (a.accum) v += a.y[addr];
        if (a.relu && v < 0.f) v = 0.f;
        a.y[addr] = v;
      }
    }
  }
}

template <int KK>
__global__ __launch_bounds__(32) void conv_wmma_direct(ConvArgs a) {
  constexpr int KSQ = KK * KK;
  const int lane = threadIdx.x;
  const int half = lane >> 4;
  const int l16  = lane & 15;

  int z  = blockIdx.z;
  int i2 = z % a.B2;  int t = z / a.B2;
  int i1 = t % a.B1;  int i0 = t / a.B1;
  const long inBase  = (long)i0 * a.inB0  + (long)i1 * a.inB1  + (long)i2 * a.inB2;
  const long outBase = (long)i0 * a.outB0 + (long)i1 * a.outB1 + (long)i2 * a.outB2;

  const int pix = blockIdx.x * 16 + l16;
  const bool pvalid = pix < a.Hout * a.Wout;
  int ho = -1, wo = 0;
  if (pvalid) { ho = pix / a.Wout; wo = pix - ho * a.Wout; }
  const int hbase = ho * a.stride - a.pad;
  const int wbase = wo * a.stride - a.pad;

  const int coTile = blockIdx.y * 16;
  const int m = coTile + l16;
  const __bf16* wrow = a.w + (long)m * a.Kpad;

  v8f_t acc = {};
  for (int kk = 0; kk < a.Kpad; kk += 32) {
    union { v16bf_t v; uint4 q[2]; } au;
    if (m < a.Cout) {
      au.q[0] = *(const uint4*)(wrow + kk + (half ? 16 : 0));
      au.q[1] = *(const uint4*)(wrow + kk + (half ? 24 : 8));
    } else {
      au.q[0] = make_uint4(0u, 0u, 0u, 0u);
      au.q[1] = make_uint4(0u, 0u, 0u, 0u);
    }
    union { v16bf_t v; __bf16 e[16]; } bu;
    int K0 = kk + half * 16;
    int ci = K0 / KSQ;
    int r  = K0 - ci * KSQ;
    int kh = r / KK;
    int kw = r - kh * KK;
#pragma unroll
    for (int j = 0; j < 16; ++j) {
      float v = 0.f;
      if (pvalid && ci < a.Cin) {
        int hi = hbase + kh, wi = wbase + kw;
        if (hi >= 0 && hi < a.Hin && wi >= 0 && wi < a.Win)
          v = a.x[inBase + (long)ci * a.inC + (long)hi * a.inH + (long)wi * a.inW];
      }
      bu.e[j] = (__bf16)v;
      if (++kw == KK) { kw = 0; if (++kh == KK) { kh = 0; ++ci; } }
    }
    acc = __builtin_amdgcn_wmma_f32_16x16x32_bf16(false, au.v, false, bu.v,
                                                  (short)0, acc, false, false);
  }
  conv_store(a, acc, outBase, coTile, half, ho, wo);
}

template <int KK>
__global__ __launch_bounds__(128) void conv_wmma_shared(ConvArgs a) {
  constexpr int KSQ = KK * KK;
  __shared__ __align__(16) __bf16 lb[2][1024];   // double-buffered 64x16 B tile
  __shared__ int phw[16][2];                     // (ho,wo) per pixel of tile

  const int tid  = threadIdx.x;
  const int lane = tid & 31;
  const int wave = tid >> 5;
  const int half = lane >> 4;
  const int l16  = lane & 15;

  int z  = blockIdx.z;
  int i2 = z % a.B2;  int t = z / a.B2;
  int i1 = t % a.B1;  int i0 = t / a.B1;
  const long inBase  = (long)i0 * a.inB0  + (long)i1 * a.inB1  + (long)i2 * a.inB2;
  const long outBase = (long)i0 * a.outB0 + (long)i1 * a.outB1 + (long)i2 * a.outB2;

  if (tid < 16) {
    int p = blockIdx.x * 16 + tid;
    if (p < a.Hout * a.Wout) {
      int ho = p / a.Wout;
      phw[tid][0] = ho;
      phw[tid][1] = p - ho * a.Wout;
    } else {
      phw[tid][0] = -1;
      phw[tid][1] = 0;
    }
  }
  __syncthreads();

  // Each thread builds 8 of the 1024 B elements per 64-K step. Flat index
  // f = sub*512 + lane_f*16 + j maps to K = kk + sub*32 + (lane_f>=16?16:0)+j,
  // column N = lane_f & 15.
  int koff[8], nidx[8];
#pragma unroll
  for (int i = 0; i < 8; ++i) {
    int f   = tid + 128 * i;
    int sub = f >> 9;
    int fs  = f & 511;
    int lf  = fs >> 4, j = fs & 15;
    nidx[i] = lf & 15;
    koff[i] = sub * 32 + ((lf >> 4) << 4) + j;
  }

  const int coTile = blockIdx.y * 64 + wave * 16;
  const int m = coTile + l16;
  const __bf16* wrow = a.w + (long)m * a.Kpad;

  v8f_t acc = {};
  int buf = 0;

  for (int kk = 0; kk < a.Kpad; kk += 64) {
    const bool has2 = (kk + 32) < a.Kpad;   // wave-uniform tail guard

    // ---- cooperative im2col gather into LDS (f32 load -> bf16 store)
#pragma unroll
    for (int i = 0; i < 8; ++i) {
      int K  = kk + koff[i];
      int ci = K / KSQ;
      int r  = K - ci * KSQ;
      int kh = r / KK;
      int kw = r - kh * KK;
      int ho = phw[nidx[i]][0], wo = phw[nidx[i]][1];
      float v = 0.f;
      if (ho >= 0 && ci < a.Cin) {
        int hi = ho * a.stride - a.pad + kh;
        int wi = wo * a.stride - a.pad + kw;
        if (hi >= 0 && hi < a.Hin && wi >= 0 && wi < a.Win)
          v = a.x[inBase + (long)ci * a.inC + (long)hi * a.inH + (long)wi * a.inW];
      }
      lb[buf][tid + 128 * i] = (__bf16)v;
    }

    // ---- A fragment for sub-step 0: issued before the barrier so the
    //      global-load wait overlaps barrier latency.
    union { v16bf_t v; uint4 q[2]; } a0;
    if (m < a.Cout) {
      a0.q[0] = *(const uint4*)(wrow + kk + (half ? 16 : 0));
      a0.q[1] = *(const uint4*)(wrow + kk + (half ? 24 : 8));
    } else {
      a0.q[0] = make_uint4(0u, 0u, 0u, 0u);
      a0.q[1] = make_uint4(0u, 0u, 0u, 0u);
    }

    __syncthreads();   // one barrier per 64-K step (double-buffered LDS)

    union { v16bf_t v; uint4 q[2]; } b0;
    const __bf16* s0 = &lb[buf][lane * 16];
    b0.q[0] = *(const uint4*)(s0);
    b0.q[1] = *(const uint4*)(s0 + 8);
    acc = __builtin_amdgcn_wmma_f32_16x16x32_bf16(false, a0.v, false, b0.v,
                                                  (short)0, acc, false, false);

    if (has2) {
      union { v16bf_t v; uint4 q[2]; } a1, b1;
      if (m < a.Cout) {
        a1.q[0] = *(const uint4*)(wrow + kk + 32 + (half ? 16 : 0));
        a1.q[1] = *(const uint4*)(wrow + kk + 32 + (half ? 24 : 8));
      } else {
        a1.q[0] = make_uint4(0u, 0u, 0u, 0u);
        a1.q[1] = make_uint4(0u, 0u, 0u, 0u);
      }
      const __bf16* s1 = &lb[buf][512 + lane * 16];
      b1.q[0] = *(const uint4*)(s1);
      b1.q[1] = *(const uint4*)(s1 + 8);
      acc = __builtin_amdgcn_wmma_f32_16x16x32_bf16(false, a1.v, false, b1.v,
                                                    (short)0, acc, false, false);
    }
    buf ^= 1;
  }

  conv_store(a, acc, outBase, coTile, half, phw[l16][0], phw[l16][1]);
}

// ---------------------------------------------------------------------------
// Weight fp32 -> bf16 with K padded to a multiple of 32 (zero fill).
// ---------------------------------------------------------------------------
__global__ void wcvt_kernel(const float* w, __bf16* o, int Cout, int Ktot, int Kpad) {
  long i = (long)blockIdx.x * blockDim.x + threadIdx.x;
  if (i >= (long)Cout * Kpad) return;
  int k  = (int)(i % Kpad);
  int co = (int)(i / Kpad);
  float v = (k < Ktot) ? w[(long)co * Ktot + k] : 0.f;
  o[i] = (__bf16)v;
}

// ---------------------------------------------------------------------------
// GroupNorm: deterministic 3-stage (segment partials -> reduce -> normalize).
// ---------------------------------------------------------------------------
__global__ __launch_bounds__(256) void gn_stats_kernel(const float* x, float* part,
                                                       long chunkLen, int nSeg) {
  __shared__ float sh0[256], sh1[256];
  const int tid = threadIdx.x;
  const int chunk = blockIdx.x;
  const int seg   = blockIdx.y;
  long segLen = (chunkLen + nSeg - 1) / nSeg;
  long s0 = (long)seg * segLen;
  long s1 = s0 + segLen; if (s1 > chunkLen) s1 = chunkLen;
  const float* p = x + (long)chunk * chunkLen;
  float sum = 0.f, ss = 0.f;
  for (long i = s0 + tid; i < s1; i += 256) { float v = p[i]; sum += v; ss += v * v; }
  sh0[tid] = sum; sh1[tid] = ss;
  __syncthreads();
  for (int off = 128; off > 0; off >>= 1) {
    if (tid < off) { sh0[tid] += sh0[tid + off]; sh1[tid] += sh1[tid + off]; }
    __syncthreads();
  }
  if (tid == 0) {
    part[((long)chunk * nSeg + seg) * 2 + 0] = sh0[0];
    part[((long)chunk * nSeg + seg) * 2 + 1] = sh1[0];
  }
}

__global__ __launch_bounds__(256) void gn_reduce_kernel(const float* part, float* out, int nSeg) {
  __shared__ float sh0[256], sh1[256];
  const int tid = threadIdx.x, chunk = blockIdx.x;
  float s = 0.f, q = 0.f;
  for (int i = tid; i < nSeg; i += 256) {
    s += part[((long)chunk * nSeg + i) * 2 + 0];
    q += part[((long)chunk * nSeg + i) * 2 + 1];
  }
  sh0[tid] = s; sh1[tid] = q;
  __syncthreads();
  for (int off = 128; off > 0; off >>= 1) {
    if (tid < off) { sh0[tid] += sh0[tid + off]; sh1[tid] += sh1[tid + off]; }
    __syncthreads();
  }
  if (tid == 0) { out[chunk * 2] = sh0[0]; out[chunk * 2 + 1] = sh1[0]; }
}

__global__ __launch_bounds__(256) void gn_norm_kernel(float* x, const float* stats,
                                                      const float* gw, const float* gb,
                                                      long S, int C, int CoverG, int G,
                                                      float invCnt, float eps, int relu) {
  int c = blockIdx.y, b = blockIdx.z;
  int chunk = b * G + c / CoverG;
  float mean = stats[2 * chunk] * invCnt;
  float var  = stats[2 * chunk + 1] * invCnt - mean * mean;
  float sc = rsqrtf(var + eps) * gw[c];
  float sh = gb[c];
  float* p = x + ((long)b * C + c) * S;
  long stride = (long)gridDim.x * 256;
  for (long i = (long)blockIdx.x * 256 + threadIdx.x; i < S; i += stride) {
    float v = (p[i] - mean) * sc + sh;
    if (relu && v < 0.f) v = 0.f;
    p[i] = v;
  }
}

// ---------------------------------------------------------------------------
// Bilinear (align_corners=True) resize over axes (2,3) of a (B,C,H,W,sb)
// tensor, with optional fused residual add.
// ---------------------------------------------------------------------------
__global__ void interp_add_kernel(const float* x, const float* add, float* y,
                                  int C, int Hi, int Wi, int Ho, int Wo, int sb,
                                  long total, float ry, float rx) {
  long idx = (long)blockIdx.x * blockDim.x + threadIdx.x;
  if (idx >= total) return;
  long t = idx;
  int s  = (int)(t % sb); t /= sb;
  int ox = (int)(t % Wo); t /= Wo;
  int oy = (int)(t % Ho); t /= Ho;
  int c  = (int)(t % C);  t /= C;
  int b  = (int)t;
  float sy = oy * ry, sx = ox * rx;
  int y0 = (int)sy; if (y0 > Hi - 2) y0 = Hi - 2; if (y0 < 0) y0 = 0;
  int x0 = (int)sx; if (x0 > Wi - 2) x0 = Wi - 2; if (x0 < 0) x0 = 0;
  float wy = sy - y0, wx = sx - x0;
  long base = ((long)b * C + c) * Hi;
  const float* p00 = x + ((base + y0) * Wi + x0) * (long)sb + s;
  float v00 = p00[0];
  float v01 = p00[sb];
  float v10 = p00[(long)Wi * sb];
  float v11 = p00[(long)Wi * sb + sb];
  float v = v00 * (1.f - wy) * (1.f - wx) + v01 * (1.f - wy) * wx
          + v10 * wy * (1.f - wx)        + v11 * wy * wx;
  if (add) v += add[idx];
  y[idx] = v;
}

// Mean over the trailing contiguous (hb*wb) support dims.
__global__ void mean_support_kernel(const float* x, float* y, long total, int sb, float inv) {
  long idx = (long)blockIdx.x * blockDim.x + threadIdx.x;
  if (idx >= total) return;
  const float* p = x + idx * (long)sb;
  float s = 0.f;
  for (int i = 0; i < sb; ++i) s += p[i];
  y[idx] = s * inv;
}

// ---------------------------------------------------------------------------
// Host orchestration
// ---------------------------------------------------------------------------
static inline int ceilDivH(int a, int b) { return (a + b - 1) / b; }
static inline int ceil32H(int a) { return (a + 31) & ~31; }

struct T6 { int b, c, ha, wa, hb, wb; };
struct LayerP { const float *w1, *b1, *w2, *b2, *gw, *gb; };

extern "C" void kernel_launch(void* const* d_in, const int* in_sizes, int n_in,
                              void* d_out, int out_size, void* d_ws, size_t ws_size,
                              hipStream_t stream) {
  (void)in_sizes; (void)n_in; (void)out_size; (void)ws_size;

  const float* corr4 = (const float*)d_in[0];
  const float* corr3 = (const float*)d_in[1];
  const float* corr2 = (const float*)d_in[2];

  int pi = 3;
  auto nextLayer = [&]() {
    LayerP L;
    L.w1 = (const float*)d_in[pi++]; L.b1 = (const float*)d_in[pi++];
    L.w2 = (const float*)d_in[pi++]; L.b2 = (const float*)d_in[pi++];
    L.gw = (const float*)d_in[pi++]; L.gb = (const float*)d_in[pi++];
    return L;
  };
  LayerP enc4P[3], enc3P[3], enc2P[3], enc43P[3], enc32P[3];
  for (int i = 0; i < 3; ++i) enc4P[i]  = nextLayer();
  for (int i = 0; i < 3; ++i) enc3P[i]  = nextLayer();
  for (int i = 0; i < 3; ++i) enc2P[i]  = nextLayer();
  for (int i = 0; i < 3; ++i) enc43P[i] = nextLayer();
  for (int i = 0; i < 3; ++i) enc32P[i] = nextLayer();
  const float* d1w1 = (const float*)d_in[pi++]; const float* d1b1 = (const float*)d_in[pi++];
  const float* d1w2 = (const float*)d_in[pi++]; const float* d1b2 = (const float*)d_in[pi++];
  const float* d2w1 = (const float*)d_in[pi++]; const float* d2b1 = (const float*)d_in[pi++];
  const float* d2w2 = (const float*)d_in[pi++]; const float* d2b2 = (const float*)d_in[pi++];

  // ---- workspace bump allocator ----
  char* wsb = (char*)d_ws;
  size_t off = 0;
  auto allocB = [&](size_t bytes) -> void* {
    void* p = wsb + off;
    off = (off + bytes + 255) & ~(size_t)255;
    return p;
  };

  // ---- bf16 weight conversion ----
  auto cvtW = [&](const float* w, int co, int ci, int k) -> const __bf16* {
    int Ktot = ci * k * k, Kpad = ceil32H(Ktot);
    __bf16* p = (__bf16*)allocB((size_t)co * Kpad * sizeof(__bf16));
    long n = (long)co * Kpad;
    wcvt_kernel<<<dim3((unsigned)((n + 255) / 256)), dim3(256), 0, stream>>>(w, p, co, Ktot, Kpad);
    return p;
  };

  int encCh[3] = {16, 64, 128};
  const __bf16 *W4[3][2], *W3[3][2], *W2[3][2], *W43[3][2], *W32[3][2];
  {
    int k4[3] = {3, 3, 3}, k3_[3] = {5, 3, 3}, k2_[3] = {5, 5, 3};
    int ci = 3;
    for (int i = 0; i < 3; ++i) { W4[i][0] = cvtW(enc4P[i].w1, encCh[i], ci, k4[i]);
                                  W4[i][1] = cvtW(enc4P[i].w2, encCh[i], ci, k4[i]); ci = encCh[i]; }
    ci = 6;
    for (int i = 0; i < 3; ++i) { W3[i][0] = cvtW(enc3P[i].w1, encCh[i], ci, k3_[i]);
                                  W3[i][1] = cvtW(enc3P[i].w2, encCh[i], ci, k3_[i]); ci = encCh[i]; }
    ci = 4;
    for (int i = 0; i < 3; ++i) { W2[i][0] = cvtW(enc2P[i].w1, encCh[i], ci, k2_[i]);
                                  W2[i][1] = cvtW(enc2P[i].w2, encCh[i], ci, k2_[i]); ci = encCh[i]; }
    for (int i = 0; i < 3; ++i) { W43[i][0] = cvtW(enc43P[i].w1, 128, 128, 3);
                                  W43[i][1] = cvtW(enc43P[i].w2, 128, 128, 3); }
    for (int i = 0; i < 3; ++i) { W32[i][0] = cvtW(enc32P[i].w1, 128, 128, 3);
                                  W32[i][1] = cvtW(enc32P[i].w2, 128, 128, 3); }
  }
  const __bf16* Wd1a = cvtW(d1w1, 128, 128, 3);
  const __bf16* Wd1b = cvtW(d1w2,  64, 128, 3);
  const __bf16* Wd2a = cvtW(d2w1,  64,  64, 3);
  const __bf16* Wd2b = cvtW(d2w2,   2,  64, 3);

  // GN scratch (deterministic partials)
  const int MAXSEG = 256;
  float* gnPart  = (float*)allocB((size_t)16 * MAXSEG * 2 * sizeof(float));
  float* gnStats = (float*)allocB(32 * sizeof(float));

  auto numel = [](T6 d) { return (size_t)d.b * d.c * d.ha * d.wa * d.hb * d.wb; };

  // persistent per-scale results
  T6 s4d{4, 128, 13, 13, 2, 2}, s3d{4, 128, 26, 26, 2, 2}, s2d{4, 128, 52, 52, 2, 2};
  float* s4res = (float*)allocB(numel(s4d) * sizeof(float));
  float* s3res = (float*)allocB(numel(s3d) * sizeof(float));
  float* s2res = (float*)allocB(numel(s2d) * sizeof(float));
  float* scratch = (float*)(wsb + off);   // remainder of ws, reused per stage

  // ---- conv launch with template dispatch (k in {3,5}; shared for Cout>=64) ----
  auto convLaunch = [&](ConvArgs& a, int B0) {
    bool shared = a.Cout >= 64;
    int wpb = shared ? 4 : 1;
    dim3 grid(ceilDivH(a.Hout * a.Wout, 16), ceilDivH(a.Cout, 16 * wpb),
              (unsigned)(B0 * a.B1 * a.B2));
    dim3 blk(32 * wpb);
    if (a.Kk == 3) {
      if (shared) conv_wmma_shared<3><<<grid, blk, 0, stream>>>(a);
      else        conv_wmma_direct<3><<<grid, blk, 0, stream>>>(a);
    } else {
      if (shared) conv_wmma_shared<5><<<grid, blk, 0, stream>>>(a);
      else        conv_wmma_direct<5><<<grid, blk, 0, stream>>>(a);
    }
  };

  auto cpConv = [&](const float* x, T6 in, float* y, int co, int k, int s,
                    const __bf16* w1, const float* b1, const __bf16* w2, const float* b2) -> T6 {
    int pad = k / 2;
    int hb2 = (in.hb + s - 1) / s, wb2 = (in.wb + s - 1) / s;
    T6 out{in.b, co, in.ha, in.wa, hb2, wb2};
    long i_wb = 1, i_hb = in.wb, i_wa = (long)in.hb * in.wb, i_ha = (long)in.wa * i_wa,
         i_c = (long)in.ha * i_ha, i_b = (long)in.c * i_c;
    long o_wb = 1, o_hb = wb2, o_wa = (long)hb2 * wb2, o_ha = (long)out.wa * o_wa,
         o_c = (long)out.ha * o_ha, o_b = (long)co * o_c;
    int Ktot = in.c * k * k, Kpad = ceil32H(Ktot);

    // o1: conv over (ha,wa), batch (b, hb2, wb2) with support subsample stride s
    ConvArgs a{};
    a.x = x; a.w = w1; a.bias = b1; a.y = y;
    a.B1 = hb2; a.B2 = wb2;
    a.inB0 = i_b; a.inB1 = (long)s * i_hb; a.inB2 = (long)s * i_wb;
    a.outB0 = o_b; a.outB1 = o_hb; a.outB2 = o_wb;
    a.Hin = in.ha; a.Win = in.wa; a.inH = i_ha; a.inW = i_wa;
    a.Cin = in.c; a.inC = i_c;
    a.Hout = out.ha; a.Wout = out.wa; a.outH = o_ha; a.outW = o_wa;
    a.Cout = co; a.outC = o_c;
    a.Kk = k; a.stride = 1; a.pad = pad; a.Ktot = Ktot; a.Kpad = Kpad;
    a.accum = 0; a.relu = 0;
    convLaunch(a, in.b);

    // o2: strided conv over (hb,wb), batch (b, ha, wa), accumulate into y
    ConvArgs c2{};
    c2.x = x; c2.w = w2; c2.bias = b2; c2.y = y;
    c2.B1 = in.ha; c2.B2 = in.wa;
    c2.inB0 = i_b; c2.inB1 = i_ha; c2.inB2 = i_wa;
    c2.outB0 = o_b; c2.outB1 = o_ha; c2.outB2 = o_wa;
    c2.Hin = in.hb; c2.Win = in.wb; c2.inH = i_hb; c2.inW = i_wb;
    c2.Cin = in.c; c2.inC = i_c;
    c2.Hout = hb2; c2.Wout = wb2; c2.outH = o_hb; c2.outW = o_wb;
    c2.Cout = co; c2.outC = o_c;
    c2.Kk = k; c2.stride = s; c2.pad = pad; c2.Ktot = Ktot; c2.Kpad = Kpad;
    c2.accum = 1; c2.relu = 0;
    convLaunch(c2, in.b);
    return out;
  };

  auto groupNorm = [&](float* xbuf, T6 d, const float* gw, const float* gb) {
    const int G = 4;
    long S = (long)d.ha * d.wa * d.hb * d.wb;
    long chunkLen = (long)(d.c / G) * S;
    int nChunks = d.b * G;
    int nSeg = (int)((chunkLen + 16384 - 1) / 16384);
    if (nSeg < 1) nSeg = 1; if (nSeg > MAXSEG) nSeg = MAXSEG;
    gn_stats_kernel<<<dim3(nChunks, nSeg), dim3(256), 0, stream>>>(xbuf, gnPart, chunkLen, nSeg);
    gn_reduce_kernel<<<dim3(nChunks), dim3(256), 0, stream>>>(gnPart, gnStats, nSeg);
    float invCnt = 1.f / (float)chunkLen;
    int xblocks = (int)((S + 255) / 256); if (xblocks > 1024) xblocks = 1024;
    gn_norm_kernel<<<dim3(xblocks, d.c, d.b), dim3(256), 0, stream>>>(
        xbuf, gnStats, gw, gb, S, d.c, d.c / G, G, invCnt, 1e-5f, 1);
  };

  auto interpAdd = [&](const float* x, const float* add, float* y, T6 in, int Ho, int Wo) -> T6 {
    int sb = in.hb * in.wb;
    long total = (long)in.b * in.c * Ho * Wo * sb;
    float ry = (float)(in.ha - 1) / (float)(Ho - 1);
    float rx = (float)(in.wa - 1) / (float)(Wo - 1);
    interp_add_kernel<<<dim3((unsigned)((total + 255) / 256)), dim3(256), 0, stream>>>(
        x, add, y, in.c, in.ha, in.wa, Ho, Wo, sb, total, ry, rx);
    return T6{in.b, in.c, Ho, Wo, in.hb, in.wb};
  };

  auto runEnc = [&](const float* x, T6 in, LayerP* L, const __bf16* (*W)[2],
                    const int* ks, const int* ss, float* scr, float* res) -> T6 {
    T6 d0{in.b, encCh[0], in.ha, in.wa, (in.hb + ss[0] - 1) / ss[0], (in.wb + ss[0] - 1) / ss[0]};
    float* t0 = scr;
    float* t1 = scr + numel(d0);
    float* outs[3] = {t0, t1, res};
    T6 d = in;
    const float* cur = x;
    for (int i = 0; i < 3; ++i) {
      d = cpConv(cur, d, outs[i], encCh[i], ks[i], ss[i], W[i][0], L[i].b1, W[i][1], L[i].b2);
      groupNorm(outs[i], d, L[i].gw, L[i].gb);
      cur = outs[i];
    }
    return d;
  };

  auto conv2d = [&](const float* x, float* y, int B, int Ci, int H, int Wd,
                    const __bf16* w, const float* bias, int Co, int k, int relu) {
    ConvArgs a{};
    int pad = k / 2;
    a.x = x; a.w = w; a.bias = bias; a.y = y;
    a.B1 = 1; a.B2 = 1;
    a.inB0 = (long)Ci * H * Wd; a.inB1 = 0; a.inB2 = 0;
    a.outB0 = (long)Co * H * Wd; a.outB1 = 0; a.outB2 = 0;
    a.Hin = H; a.Win = Wd; a.inH = Wd; a.inW = 1;
    a.Cin = Ci; a.inC = (long)H * Wd;
    a.Hout = H; a.Wout = Wd; a.outH = Wd; a.outW = 1;
    a.Cout = Co; a.outC = (long)H * Wd;
    a.Kk = k; a.stride = 1; a.pad = pad;
    a.Ktot = Ci * k * k; a.Kpad = ceil32H(a.Ktot);
    a.accum = 0; a.relu = relu;
    convLaunch(a, B);
  };

  // ================= pipeline =================
  {
    int k4[3] = {3, 3, 3}, s4c[3] = {2, 2, 2};
    runEnc(corr4, T6{4, 3, 13, 13, 13, 13}, enc4P, W4, k4, s4c, scratch, s4res);
  }
  {
    int k3_[3] = {5, 3, 3}, s3c[3] = {4, 2, 2};
    runEnc(corr3, T6{4, 6, 26, 26, 26, 26}, enc3P, W3, k3_, s3c, scratch, s3res);
  }
  {
    int k2_[3] = {5, 5, 3}, s2c[3] = {4, 4, 2};
    runEnc(corr2, T6{4, 4, 52, 52, 52, 52}, enc2P, W2, k2_, s2c, scratch, s2res);
  }

  size_t szMix26 = numel(s3d);    // 1,384,448
  size_t szMix52 = numel(s2d);    // 5,537,792
  float* A  = scratch;
  float* Bb = A  + szMix26;
  float* Cc = Bb + szMix26;
  float* D  = Cc + szMix26;
  float* E  = D  + szMix52;
  float* F  = E  + szMix52;
  float* G  = F  + szMix52;                       // 4*128*52*52
  float* H  = G  + (size_t)4 * 128 * 52 * 52;
  float* I  = H  + (size_t)4 * 128 * 52 * 52;     // 4*64*104*104
  float* J  = I  + (size_t)4 * 64 * 104 * 104;

  // mix43: interp(s4) + s3 -> 3 cp-conv layers
  T6 d = interpAdd(s4res, s3res, A, s4d, 26, 26);
  {
    float* rot[4] = {A, Bb, Cc, A};
    for (int i = 0; i < 3; ++i) {
      d = cpConv(rot[i], d, rot[i + 1], 128, 3, 1, W43[i][0], enc43P[i].b1, W43[i][1], enc43P[i].b2);
      groupNorm(rot[i + 1], d, enc43P[i].gw, enc43P[i].gb);
    }
  }
  // mix32: interp(m43) + s2 -> 3 cp-conv layers
  d = interpAdd(A, s2res, D, d, 52, 52);
  {
    float* rot[4] = {D, E, F, D};
    for (int i = 0; i < 3; ++i) {
      d = cpConv(rot[i], d, rot[i + 1], 128, 3, 1, W32[i][0], enc32P[i].b1, W32[i][1], enc32P[i].b2);
      groupNorm(rot[i + 1], d, enc32P[i].gw, enc32P[i].gb);
    }
  }

  // mean over support dims -> (4,128,52,52)
  {
    long total = (long)4 * 128 * 52 * 52;
    mean_support_kernel<<<dim3((unsigned)((total + 255) / 256)), dim3(256), 0, stream>>>(
        D, G, total, 4, 0.25f);
  }

  // decoder
  conv2d(G, H, 4, 128, 52, 52, Wd1a, d1b1, 128, 3, 1);
  conv2d(H, G, 4, 128, 52, 52, Wd1b, d1b2, 64, 3, 1);
  interpAdd(G, nullptr, I, T6{4, 64, 52, 52, 1, 1}, 104, 104);
  conv2d(I, J, 4, 64, 104, 104, Wd2a, d2b1, 64, 3, 1);
  conv2d(J, (float*)d_out, 4, 64, 104, 104, Wd2b, d2b2, 2, 3, 0);
}
